// SparseLowrankAttention_22265110463161
// MI455X (gfx1250) — matrix operand order repaired and checked by reading, not compiled, not run on hardware
//
#include <hip/hip_runtime.h>

// B=2, H=8, S=2048, D=64, DIAG=64 (from reference setup)
#define S_LEN 2048
#define D_DIM 64
#define DIAG  64
#define KV_SPLIT 8   // S-dimension split for the kv reduction (256 rows each)

typedef __attribute__((ext_vector_type(16))) __bf16 v16bf;
typedef __attribute__((ext_vector_type(8)))  float  v8f;

#define WMMA_BF16(a, b, c) \
  __builtin_amdgcn_wmma_f32_16x16x32_bf16(false, (a), false, (b), (short)0, (c), false, false)

__device__ __forceinline__ unsigned short f2bf(float f) {
  union { float f; unsigned u; } c; c.f = f;
  unsigned u = c.u;
  return (unsigned short)((u + 0x7FFFu + ((u >> 16) & 1u)) >> 16);  // RNE
}

// A fragment, 16x32 bf16 (ISA 7.12.2): lane L -> m=L&15, half=L>>4;
// VGPR 0-3: k = half*8 + {0..7}, VGPR 4-7: k = 16 + half*8 + {0..7}
__device__ __forceinline__ v16bf load_fragA(const unsigned short* base, int ld, int lane) {
  int r = lane & 15, h = lane >> 4;
  union { v16bf v; uint4 u[2]; } f;
  f.u[0] = *(const uint4*)(base + r * ld + h * 8);
  f.u[1] = *(const uint4*)(base + r * ld + 16 + h * 8);
  return f.v;
}

// B fragment, 32x16 bf16: lane L -> n=L&15, half=L>>4;
// VGPR 0-3: k = half*8 + {0..7}, VGPR 4-7: k = 16 + half*8 + {0..7}. Operand stored [n][k] in LDS.
__device__ __forceinline__ v16bf load_fragB(const unsigned short* base, int ld, int lane) {
  int r = lane & 15, h = lane >> 4;
  union { v16bf v; uint4 u[2]; } f;
  f.u[0] = *(const uint4*)(base + r * ld + h * 8);
  f.u[1] = *(const uint4*)(base + r * ld + 16 + h * 8);
  return f.v;
}

// ------------- Kernel 1a: partial kv = K1^T @ Vm over a 256-row S-chunk -------------
__global__ __launch_bounds__(128) void kv_partial_kernel(
    const float* __restrict__ K, const float* __restrict__ V,
    const float* __restrict__ mask, float* __restrict__ partial) {
  __shared__ __attribute__((aligned(16))) unsigned short K1t[64 * 32];  // [d][t]
  __shared__ __attribute__((aligned(16))) unsigned short Vmt[64 * 32];  // [e][t]
  const int bh = blockIdx.x;           // 0..15
  const int sp = blockIdx.y;           // 0..KV_SPLIT-1
  const int b = bh >> 3;
  const int tid = threadIdx.x, lane = tid & 31, wave = tid >> 5;
  const float lrs = 0.14865088937534013f;  // S^-0.25
  const int tbase = sp * (S_LEN / KV_SPLIT);
  const float* Kp = K + (size_t)bh * S_LEN * D_DIM;
  const float* Vp = V + (size_t)bh * S_LEN * D_DIM;
  const float* mp = mask + (size_t)b * S_LEN;

  const v8f vzero = {0.f, 0.f, 0.f, 0.f, 0.f, 0.f, 0.f, 0.f};
  v8f acc[4] = {vzero, vzero, vzero, vzero};

  for (int c = 0; c < (S_LEN / KV_SPLIT) / 32; ++c) {
    const int t0 = tbase + c * 32;
    // prefetch the next chunk while this one computes
    if (c + 1 < (S_LEN / KV_SPLIT) / 32) {
      __builtin_prefetch(Kp + (size_t)(t0 + 32) * 64 + tid * 16, 0, 1);
      __builtin_prefetch(Vp + (size_t)(t0 + 32) * 64 + tid * 16, 0, 1);
    }
    __syncthreads();  // previous iteration's WMMA reads done
    for (int idx = tid; idx < 32 * 64; idx += 128) {
      int t = idx >> 6, d = idx & 63;
      float m = mp[t0 + t];
      float k = Kp[(size_t)(t0 + t) * 64 + d];
      float k1 = (k > 0.f ? k + 1.f : __expf(k)) * m * lrs;  // (elu+1)*mask*scale
      K1t[d * 32 + t] = f2bf(k1);
      float v = Vp[(size_t)(t0 + t) * 64 + d];
      Vmt[d * 32 + t] = f2bf(v * m);
    }
    __syncthreads();
    v16bf a = load_fragA(K1t + wave * 16 * 32, 32, lane);  // A(m=d, k=t)
#pragma unroll
    for (int n0 = 0; n0 < 4; ++n0) {
      v16bf bf = load_fragB(Vmt + n0 * 16 * 32, 32, lane);  // B(k=t, n=e)
      acc[n0] = WMMA_BF16(a, bf, acc[n0]);
    }
  }

  float* outp = partial + (size_t)(sp * 16 + bh) * 4096;
  const int n = lane & 15, h = lane >> 4;
#pragma unroll
  for (int n0 = 0; n0 < 4; ++n0)
#pragma unroll
    for (int v = 0; v < 8; ++v) {
      int d = wave * 16 + v + 8 * h;
      int e = n0 * 16 + n;
      outp[d * 64 + e] = acc[n0][v];
    }
}

// ------------- Kernel 1b: deterministic sum of the KV_SPLIT partials -------------
__global__ __launch_bounds__(256) void kv_sum_kernel(
    const float* __restrict__ partial, float* __restrict__ kv_out) {
  const int total = 16 * 4096;  // bh * (64x64)
  for (int i = blockIdx.x * 256 + threadIdx.x; i < total; i += gridDim.x * 256) {
    float s = 0.f;
#pragma unroll
    for (int sp = 0; sp < KV_SPLIT; ++sp) s += partial[(size_t)sp * total + i];
    kv_out[i] = s;
  }
}

// ------------- Kernel 2: banded attention + low-rank + blend, per 64-row tile -------------
__global__ __launch_bounds__(128) void band_attn_kernel(
    const float* __restrict__ Q, const float* __restrict__ K,
    const float* __restrict__ V, const float* __restrict__ mask,
    const float* __restrict__ ratio, const float* __restrict__ kv,
    float* __restrict__ out) {
  // 64 KB static LDS, phase-overlaid
  __shared__ __attribute__((aligned(16))) unsigned short smem[32768];
  unsigned short* Q1b = smem;          // phase1: (elu(Q)+1)*scale bf16 [64][64]
  unsigned short* kvT = smem + 4096;   // phase1: kv transposed [e][d]
  unsigned short* Qb  = smem;          // phase2: Q bf16 [64][64]
  unsigned short* Kb  = smem + 4096;   // phase2: current K tile [n][k]=[row][d]
  unsigned short* Vt  = smem + 8192;   // phase2: 3 V tiles transposed [e][t]
  unsigned short* Pb  = smem + 20480;  // phase2: probs bf16 [64][192]

  const int qt = blockIdx.x;   // 0..31
  const int bh = blockIdx.y;   // 0..15
  const int b = bh >> 3;
  const int q0 = qt * 64;
  const int tid = threadIdx.x, lane = tid & 31, wave = tid >> 5;
  const int n = lane & 15, h = lane >> 4;
  const float lrs = 0.14865088937534013f;

  const float* Qp = Q + (size_t)bh * S_LEN * 64;
  const float* Kp = K + (size_t)bh * S_LEN * 64;
  const float* Vp = V + (size_t)bh * S_LEN * 64;
  const float* mp = mask + (size_t)b * S_LEN;
  const float* kvp = kv + bh * 4096;

  const v8f vzero = {0.f, 0.f, 0.f, 0.f, 0.f, 0.f, 0.f, 0.f};

  const int  kt[3]     = {qt > 0 ? qt - 1 : 0, qt, qt < 31 ? qt + 1 : 31};
  const bool tvalid[3] = {qt > 0, true, qt < 31};

  // prefetch first attention tiles while the low-rank phase runs
  __builtin_prefetch(Kp + (size_t)kt[0] * 64 * 64 + tid * 16, 0, 1);
  __builtin_prefetch(Vp + (size_t)kt[0] * 64 * 64 + tid * 16, 0, 1);

  // ---- Phase 1: low-rank path OL = Q1 @ kv ----
  for (int idx = tid; idx < 4096; idx += 128) {
    int r = idx >> 6, cc = idx & 63;
    float q = Qp[(size_t)(q0 + r) * 64 + cc];
    Q1b[idx] = f2bf((q > 0.f ? q + 1.f : __expf(q)) * lrs);
    kvT[cc * 64 + r] = f2bf(kvp[idx]);  // kv[d=r][e=cc] -> [e][d]
  }
  __syncthreads();
  v8f OL[4];
  {
    v16bf a0 = load_fragA(Q1b + wave * 16 * 64, 64, lane);       // k = 0..31
    v16bf a1 = load_fragA(Q1b + wave * 16 * 64 + 32, 64, lane);  // k = 32..63
#pragma unroll
    for (int n0 = 0; n0 < 4; ++n0) {
      v8f a = vzero;
      v16bf b0 = load_fragB(kvT + n0 * 16 * 64, 64, lane);
      v16bf b1 = load_fragB(kvT + n0 * 16 * 64 + 32, 64, lane);
      a = WMMA_BF16(a0, b0, a);
      a = WMMA_BF16(a1, b1, a);
      OL[n0] = a;
    }
  }
  __syncthreads();  // phase-1 LDS reads done before overlay reuse

  // ---- Phase 2: banded attention ----
  for (int idx = tid; idx < 4096; idx += 128) {
    int r = idx >> 6, cc = idx & 63;
    Qb[idx] = f2bf(Qp[(size_t)(q0 + r) * 64 + cc]);
  }

  v8f Sreg[3][4];
#pragma unroll
  for (int it = 0; it < 3; ++it) {
    const int k0 = kt[it] * 64;
    if (it < 2) {  // prefetch next tile's K/V
      __builtin_prefetch(Kp + (size_t)kt[it + 1] * 64 * 64 + tid * 16, 0, 1);
      __builtin_prefetch(Vp + (size_t)kt[it + 1] * 64 * 64 + tid * 16, 0, 1);
    }
    for (int idx = tid; idx < 4096; idx += 128) {
      int r = idx >> 6, cc = idx & 63;
      Kb[idx] = f2bf(Kp[(size_t)(k0 + r) * 64 + cc]);            // [row=n][k=d]
      Vt[it * 4096 + cc * 64 + r] = f2bf(Vp[(size_t)(k0 + r) * 64 + cc]);  // [e][t]
    }
    __syncthreads();
    v16bf a0 = load_fragA(Qb + wave * 16 * 64, 64, lane);
    v16bf a1 = load_fragA(Qb + wave * 16 * 64 + 32, 64, lane);
#pragma unroll
    for (int n0 = 0; n0 < 4; ++n0) {
      v8f acc = vzero;
      v16bf b0 = load_fragB(Kb + n0 * 16 * 64, 64, lane);
      v16bf b1 = load_fragB(Kb + n0 * 16 * 64 + 32, 64, lane);
      acc = WMMA_BF16(a0, b0, acc);
      acc = WMMA_BF16(a1, b1, acc);
#pragma unroll
      for (int v = 0; v < 8; ++v) {  // scale + band mask + key mask
        int i = q0 + wave * 16 + v + 8 * h;
        int j = k0 + n0 * 16 + n;
        float s = acc[v] * 0.125f;   // 1/sqrt(D)
        int dd = i - j; if (dd < 0) dd = -dd;
        if (dd >= DIAG || !tvalid[it]) s = -1e6f;
        s -= 1e6f * (1.f - mp[j]);
        acc[v] = s;
      }
      Sreg[it][n0] = acc;
    }
    __syncthreads();  // protect Kb before next tile overwrites
  }

  // ---- softmax over 192 band columns, fully in registers ----
#pragma unroll
  for (int v = 0; v < 8; ++v) {
    float m = -1e30f;
#pragma unroll
    for (int it = 0; it < 3; ++it)
#pragma unroll
      for (int n0 = 0; n0 < 4; ++n0) m = fmaxf(m, Sreg[it][n0][v]);
    for (int mk = 1; mk < 16; mk <<= 1) m = fmaxf(m, __shfl_xor(m, mk, 32));
    float sum = 0.f;
#pragma unroll
    for (int it = 0; it < 3; ++it)
#pragma unroll
      for (int n0 = 0; n0 < 4; ++n0) {
        float e = __expf(Sreg[it][n0][v] - m);
        Sreg[it][n0][v] = e;
        sum += e;
      }
    for (int mk = 1; mk < 16; mk <<= 1) sum += __shfl_xor(sum, mk, 32);
    float inv = 1.f / sum;
#pragma unroll
    for (int it = 0; it < 3; ++it)
#pragma unroll
      for (int n0 = 0; n0 < 4; ++n0) Sreg[it][n0][v] *= inv;
  }

  // route P through LDS to convert D-frag layout -> A-frag layout
#pragma unroll
  for (int it = 0; it < 3; ++it)
#pragma unroll
    for (int n0 = 0; n0 < 4; ++n0)
#pragma unroll
      for (int v = 0; v < 8; ++v) {
        int row = wave * 16 + v + 8 * h;
        int col = it * 64 + n0 * 16 + n;
        Pb[row * 192 + col] = f2bf(Sreg[it][n0][v]);
      }
  __syncthreads();

  // ---- O = P @ V (invalid tiles contribute 0 since P==0 there) ----
  v8f O[4];
#pragma unroll
  for (int n0 = 0; n0 < 4; ++n0) {
    v8f acc = vzero;
#pragma unroll
    for (int it = 0; it < 3; ++it) {
      v16bf a0 = load_fragA(Pb + wave * 16 * 192 + it * 64, 192, lane);
      v16bf a1 = load_fragA(Pb + wave * 16 * 192 + it * 64 + 32, 192, lane);
      v16bf b0 = load_fragB(Vt + it * 4096 + n0 * 16 * 64, 64, lane);
      v16bf b1 = load_fragB(Vt + it * 4096 + n0 * 16 * 64 + 32, 64, lane);
      acc = WMMA_BF16(a0, b0, acc);
      acc = WMMA_BF16(a1, b1, acc);
    }
    O[n0] = acc;
  }

  // ---- blend + store ----
  const float r = ratio[0];
  float* op = out + (size_t)bh * S_LEN * 64;
#pragma unroll
  for (int n0 = 0; n0 < 4; ++n0)
#pragma unroll
    for (int v = 0; v < 8; ++v) {
      int row = q0 + wave * 16 + v + 8 * h;
      int col = n0 * 16 + n;
      op[(size_t)row * 64 + col] = r * O[n0][v] + (1.f - r) * OL[n0][v];
    }
}

extern "C" void kernel_launch(void* const* d_in, const int* in_sizes, int n_in,
                              void* d_out, int out_size, void* d_ws, size_t ws_size,
                              hipStream_t stream) {
  const float* Q     = (const float*)d_in[0];
  const float* K     = (const float*)d_in[1];
  const float* V     = (const float*)d_in[2];
  const float* mask  = (const float*)d_in[3];
  const float* ratio = (const float*)d_in[4];

  float* partial = (float*)d_ws;                      // KV_SPLIT*16*4096 floats = 2 MB
  float* kv      = partial + (size_t)KV_SPLIT * 16 * 4096;  // 16*4096 floats = 256 KB

  kv_partial_kernel<<<dim3(16, KV_SPLIT), dim3(128), 0, stream>>>(K, V, mask, partial);
  kv_sum_kernel<<<dim3(64), dim3(256), 0, stream>>>(partial, kv);
  band_attn_kernel<<<dim3(32, 16), dim3(128), 0, stream>>>(Q, K, V, mask, ratio, kv,
                                                           (float*)d_out);
}
